// SPGR_Unit_32925219291739
// MI455X (gfx1250) — compile-verified, matrix operand order checked by reading, not run on hardware
//
#include <hip/hip_runtime.h>
#include <hip/hip_bf16.h>

#define B_   2
#define C_   256
#define IC_  128
#define N_   4096
#define KSEL 2048

typedef __attribute__((ext_vector_type(16))) _Float16 v16h;
typedef __attribute__((ext_vector_type(8)))  float    v8f;

#define LDT 40   // ushort row stride for a 32-wide K tile (+8 pad, keeps 16B align, conflict-free)
#define BM  128
#define BN  64

__device__ __forceinline__ unsigned short f2h_bits(float f) {
    _Float16 h = (_Float16)f;
    unsigned short u;
    __builtin_memcpy(&u, &h, 2);
    return u;
}
__device__ __forceinline__ float h2f_bits(unsigned short u) {
    _Float16 h;
    __builtin_memcpy(&h, &u, 2);
    return (float)h;
}

// Build a 16xfp16 fragment from two contiguous 16B chunks in LDS.
// Elements 0..7  = K offsets  8*half .. 8*half+7
// Elements 8..15 = K offsets 16+8*half .. 16+8*half+7  (base + 16 ushorts)
__device__ __forceinline__ v16h ld_frag(const unsigned short* p0) {
    v16h r;
    ((uint4*)&r)[0] = *(const uint4*)(p0);
    ((uint4*)&r)[1] = *(const uint4*)(p0 + 16);
    return r;
}

// Canonical tile GEMM: C[128x64] += A[128xK] * B[64xK]^T, fp16 in, fp32 acc.
// 256 threads = 8 waves in 4(M) x 2(N); each wave owns a 32x32 macro-tile
// (2x2 WMMA 16x16x32 -> 4 WMMAs per K-step from 4 fragment loads).
// Double-buffered LDS, one barrier per K-step; next tile's global loads are
// issued before the WMMA burst so load latency overlaps compute.
__device__ __forceinline__ void gemm128x64(
    const unsigned short* __restrict__ A, int lda,
    const unsigned short* __restrict__ Bm, int ldb,
    int K, int m0, int n0,
    unsigned short* As,      // 2 * BM * LDT ushorts
    unsigned short* Bs,      // 2 * BN * LDT ushorts
    v8f acc[2][2])
{
    const int tid  = threadIdx.x;
    const int lane = tid & 31;
    const int wid  = tid >> 5;
    const int wm   = wid >> 1;          // 0..3
    const int wn   = wid & 1;           // 0..1
    const int r    = lane & 15;
    const int hf   = lane >> 4;
    // A staging: row = tid>>1 (0..127), two 8-halfword chunks at (tid&1)*16 (+8)
    const int arow = tid >> 1, acol = (tid & 1) * 16;
    // B staging: row = tid>>2 (0..63), one chunk at (tid&3)*8
    const int brow = tid >> 2, bcol = (tid & 3) * 8;

    const unsigned short* Aptr = &A[(size_t)(m0 + arow) * lda + acol];
    const unsigned short* Bptr = &Bm[(size_t)(n0 + brow) * ldb + bcol];

    // prologue: stage tile 0 into buffer 0
    uint4 ra0 = *(const uint4*)(Aptr);
    uint4 ra1 = *(const uint4*)(Aptr + 8);
    uint4 rb  = *(const uint4*)(Bptr);
    int buf = 0;
    *(uint4*)&As[arow * LDT + acol]     = ra0;
    *(uint4*)&As[arow * LDT + acol + 8] = ra1;
    *(uint4*)&Bs[brow * LDT + bcol]     = rb;
    __syncthreads();

    for (int k0 = 0; k0 < K; k0 += 32) {
        const bool more = (k0 + 32) < K;
        if (more) {
            ra0 = *(const uint4*)(Aptr + (k0 + 32));
            ra1 = *(const uint4*)(Aptr + (k0 + 32) + 8);
            rb  = *(const uint4*)(Bptr + (k0 + 32));
            __builtin_prefetch(Aptr + (k0 + 64), 0, 1);
            __builtin_prefetch(Bptr + (k0 + 64), 0, 1);
        }
        const unsigned short* Ab = &As[buf * (BM * LDT)];
        const unsigned short* Bb = &Bs[buf * (BN * LDT)];
        v16h afr[2], bfr[2];
#pragma unroll
        for (int s = 0; s < 2; ++s) {
            afr[s] = ld_frag(&Ab[(wm * 32 + s * 16 + r) * LDT + 8 * hf]);
            bfr[s] = ld_frag(&Bb[(wn * 32 + s * 16 + r) * LDT + 8 * hf]);
        }
#pragma unroll
        for (int sm = 0; sm < 2; ++sm)
#pragma unroll
            for (int sn = 0; sn < 2; ++sn)
                acc[sm][sn] = __builtin_amdgcn_wmma_f32_16x16x32_f16(
                    false, afr[sm], false, bfr[sn], (short)0, acc[sm][sn], false, false);
        if (more) {
            buf ^= 1;
            *(uint4*)&As[buf * (BM * LDT) + arow * LDT + acol]     = ra0;
            *(uint4*)&As[buf * (BM * LDT) + arow * LDT + acol + 8] = ra1;
            *(uint4*)&Bs[buf * (BN * LDT) + brow * LDT + bcol]     = rb;
        }
        __syncthreads();
    }
}

// epilogue index helpers: row = m0 + wm*32 + sm*16 + hf*8 + i
//                         col = n0 + wn*32 + sn*16 + r

// ---------------- small prep kernels ----------------

__global__ void k_convert(const float* __restrict__ src, unsigned short* __restrict__ dst, int n) {
    int i = blockIdx.x * blockDim.x + threadIdx.x;
    if (i < n) dst[i] = f2h_bits(src[i]);
}

// Wcomb[j][c] = sum_i Wgcn[i][j] * Wg[i][c];  bcomb[j] = sum_i bg[i]*Wgcn[i][j]
__global__ void k_wcomb(const float* __restrict__ Wg, const float* __restrict__ Wgcn,
                        const float* __restrict__ bg,
                        unsigned short* __restrict__ WcombH, float* __restrict__ bcomb) {
    int idx = blockIdx.x * blockDim.x + threadIdx.x;   // 0..32767
    int j = idx >> 8;
    int c = idx & 255;
    float s = 0.f;
    for (int i = 0; i < IC_; ++i) s += Wgcn[i * IC_ + j] * Wg[i * C_ + c];
    WcombH[j * C_ + c] = f2h_bits(s);
    if (c == 0) {
        float b = 0.f;
        for (int i = 0; i < IC_; ++i) b += bg[i] * Wgcn[i * IC_ + j];
        bcomb[j] = b;
    }
}

// x (b, C, N) f32 -> xT (b, N, C) fp16 via LDS 32x32 tiles
__global__ void k_transpose(const float* __restrict__ x, unsigned short* __restrict__ xT) {
    __shared__ float tile[32][33];
    int b  = blockIdx.z;
    int n0 = blockIdx.x * 32;
    int c0 = blockIdx.y * 32;
    int tx = threadIdx.x & 31;
    int ty = threadIdx.x >> 5;
#pragma unroll
    for (int j = 0; j < 4; ++j) {
        int c = c0 + ty + 8 * j;
        tile[ty + 8 * j][tx] = x[((size_t)b * C_ + c) * N_ + n0 + tx];
    }
    __syncthreads();
#pragma unroll
    for (int j = 0; j < 4; ++j) {
        int n = n0 + ty + 8 * j;
        xT[((size_t)b * N_ + n) * C_ + c0 + tx] = f2h_bits(tile[tx][ty + 8 * j]);
    }
}

// ---------------- GEMM kernels ----------------

// projections: out(node, ic) = xT(node, c) dot W(ic, c) + bias[ic]
// p=0 theta (node-major), p=1 phi (node-major), p=2 combined g (stored transposed: (ic, node))
__global__ void k_proj(const unsigned short* __restrict__ xT,
                       const unsigned short* __restrict__ WthH,
                       const unsigned short* __restrict__ WphH,
                       const unsigned short* __restrict__ WcombH,
                       const float* __restrict__ bth, const float* __restrict__ bph,
                       const float* __restrict__ bcomb,
                       unsigned short* __restrict__ thetaN,
                       unsigned short* __restrict__ phiN,
                       unsigned short* __restrict__ gT) {
    __shared__ __align__(16) unsigned short As[2 * BM * LDT];
    __shared__ __align__(16) unsigned short Bs[2 * BN * LDT];
    int zb = blockIdx.z, b = zb / 3, p = zb % 3;
    const unsigned short* W = (p == 0) ? WthH : ((p == 1) ? WphH : WcombH);
    const float* bias = (p == 0) ? bth : ((p == 1) ? bph : bcomb);
    int m0 = blockIdx.y * BM, n0 = blockIdx.x * BN;
    v8f acc[2][2] = {};
    gemm128x64(xT + (size_t)b * N_ * C_, C_, W, C_, C_, m0, n0, As, Bs, acc);

    int lane = threadIdx.x & 31, wid = threadIdx.x >> 5;
    int wm = wid >> 1, wn = wid & 1, r = lane & 15, hf = lane >> 4;
#pragma unroll
    for (int sm = 0; sm < 2; ++sm)
#pragma unroll
        for (int sn = 0; sn < 2; ++sn)
#pragma unroll
            for (int i = 0; i < 8; ++i) {
                int row = m0 + wm * 32 + sm * 16 + hf * 8 + i;  // node
                int col = n0 + wn * 32 + sn * 16 + r;           // ic
                unsigned short hv = f2h_bits(acc[sm][sn][i] + bias[col]);
                if (p == 0)      thetaN[((size_t)b * N_ + row) * IC_ + col] = hv;
                else if (p == 1) phiN  [((size_t)b * N_ + row) * IC_ + col] = hv;
                else             gT    [((size_t)b * IC_ + col) * N_ + row] = hv;
            }
}

// logits: att[n][m] = theta(n,:) dot phi(m,:)
__global__ void k_att(const unsigned short* __restrict__ thetaN,
                      const unsigned short* __restrict__ phiN,
                      unsigned short* __restrict__ att) {
    __shared__ __align__(16) unsigned short As[2 * BM * LDT];
    __shared__ __align__(16) unsigned short Bs[2 * BN * LDT];
    int b = blockIdx.z;
    int m0 = blockIdx.y * BM, n0 = blockIdx.x * BN;
    v8f acc[2][2] = {};
    gemm128x64(thetaN + (size_t)b * N_ * IC_, IC_,
               phiN   + (size_t)b * N_ * IC_, IC_, IC_, m0, n0, As, Bs, acc);

    int lane = threadIdx.x & 31, wid = threadIdx.x >> 5;
    int wm = wid >> 1, wn = wid & 1, r = lane & 15, hf = lane >> 4;
#pragma unroll
    for (int sm = 0; sm < 2; ++sm)
#pragma unroll
        for (int sn = 0; sn < 2; ++sn)
#pragma unroll
            for (int i = 0; i < 8; ++i) {
                int row = m0 + wm * 32 + sm * 16 + hf * 8 + i;
                int col = n0 + wn * 32 + sn * 16 + r;
                att[((size_t)b * N_ + row) * N_ + col] = f2h_bits(acc[sm][sn][i]);
            }
}

// one block per row: softmax + exact top-k (16-bit radix select) + hard mask + renorm,
// rewrites att in place as the final normalized adjacency A (fp16).
__global__ void k_select(unsigned short* __restrict__ att, const float* __restrict__ hard_map) {
    __shared__ unsigned short vals[N_];
    __shared__ unsigned int hist[256];
    __shared__ float redf[256];
    __shared__ float s_max, s_inv;
    __shared__ int s_hb, s_rank, s_thr;

    int tid = threadIdx.x;
    int rowi = blockIdx.x, b = blockIdx.y;
    size_t base = ((size_t)b * N_ + rowi) * N_;

    float lmax = -3.0e38f;
#pragma unroll
    for (int e = 0; e < 16; ++e) {
        int idx = tid + 256 * e;
        unsigned short h = att[base + idx];
        vals[idx] = h;
        lmax = fmaxf(lmax, h2f_bits(h));
    }
    redf[tid] = lmax;
    __syncthreads();
    for (int st = 128; st > 0; st >>= 1) {
        if (tid < st) redf[tid] = fmaxf(redf[tid], redf[tid + st]);
        __syncthreads();
    }
    if (tid == 0) s_max = redf[0];

    // pass 1: histogram of high byte of order-preserving key
    hist[tid] = 0;
    __syncthreads();
#pragma unroll
    for (int e = 0; e < 16; ++e) {
        unsigned short h = vals[tid + 256 * e];
        unsigned key = h ^ ((h & 0x8000u) ? 0xFFFFu : 0x8000u);
        atomicAdd(&hist[key >> 8], 1u);
    }
    __syncthreads();
    if (tid == 0) {
        unsigned acc = 0; int hb = 0, rank = KSEL;
        for (int v = 255; v >= 0; --v) {
            unsigned nacc = acc + hist[v];
            if (nacc >= (unsigned)KSEL) { hb = v; rank = KSEL - (int)acc; break; }
            acc = nacc;
        }
        s_hb = hb; s_rank = rank;
    }
    __syncthreads();
    int hb = s_hb, rank = s_rank;

    // pass 2: low byte within the boundary bucket
    hist[tid] = 0;
    __syncthreads();
#pragma unroll
    for (int e = 0; e < 16; ++e) {
        unsigned short h = vals[tid + 256 * e];
        unsigned key = h ^ ((h & 0x8000u) ? 0xFFFFu : 0x8000u);
        if ((int)(key >> 8) == hb) atomicAdd(&hist[key & 255u], 1u);
    }
    __syncthreads();
    if (tid == 0) {
        unsigned acc = 0; int lb = 0;
        for (int v = 255; v >= 0; --v) {
            acc += hist[v];
            if ((int)acc >= rank) { lb = v; break; }
        }
        s_thr = (hb << 8) | lb;   // key of k-th largest; keep key >= thr (ties kept)
    }
    __syncthreads();
    int thr = s_thr;
    float mx = s_max;

    // sums: S_all = sum exp(v-max), S_kept = sum over kept
    float sall = 0.f, skept = 0.f;
#pragma unroll
    for (int e = 0; e < 16; ++e) {
        unsigned short h = vals[tid + 256 * e];
        unsigned key = h ^ ((h & 0x8000u) ? 0xFFFFu : 0x8000u);
        float p = __expf(h2f_bits(h) - mx);
        sall += p;
        if ((int)key >= thr) skept += p;
    }
    redf[tid] = sall; __syncthreads();
    for (int st = 128; st > 0; st >>= 1) { if (tid < st) redf[tid] += redf[tid + st]; __syncthreads(); }
    float Sall = redf[0];
    __syncthreads();
    redf[tid] = skept; __syncthreads();
    for (int st = 128; st > 0; st >>= 1) { if (tid < st) redf[tid] += redf[tid + st]; __syncthreads(); }
    if (tid == 0) {
        float Skept = redf[0];
        float hard = hard_map[(size_t)b * N_ + rowi] > 0.5f ? 1.f : 0.f;
        // A_i = (p_i/S)/(kept/S + 1e-6) * hard = p_i * hard / (Skept + 1e-6*Sall)
        s_inv = hard / (Skept + 1e-6f * Sall);
    }
    __syncthreads();
    float inv = s_inv;
#pragma unroll
    for (int e = 0; e < 16; ++e) {
        int idx = tid + 256 * e;
        unsigned short h = vals[idx];
        unsigned key = h ^ ((h & 0x8000u) ? 0xFFFFu : 0x8000u);
        float p = ((int)key >= thr) ? __expf(h2f_bits(h) - mx) * inv : 0.f;
        att[base + idx] = f2h_bits(p);
    }
}

// gc[n][j] = relu( A(n,:) dot gT(j,:) ) ; R stored (node, ic)
__global__ void k_gcn(const unsigned short* __restrict__ att,
                      const unsigned short* __restrict__ gT,
                      unsigned short* __restrict__ R) {
    __shared__ __align__(16) unsigned short As[2 * BM * LDT];
    __shared__ __align__(16) unsigned short Bs[2 * BN * LDT];
    int b = blockIdx.z;
    int m0 = blockIdx.y * BM, n0 = blockIdx.x * BN;
    v8f acc[2][2] = {};
    gemm128x64(att + (size_t)b * N_ * N_, N_,
               gT  + (size_t)b * IC_ * N_, N_, N_, m0, n0, As, Bs, acc);

    int lane = threadIdx.x & 31, wid = threadIdx.x >> 5;
    int wm = wid >> 1, wn = wid & 1, r = lane & 15, hf = lane >> 4;
#pragma unroll
    for (int sm = 0; sm < 2; ++sm)
#pragma unroll
        for (int sn = 0; sn < 2; ++sn)
#pragma unroll
            for (int i = 0; i < 8; ++i) {
                int row = m0 + wm * 32 + sm * 16 + hf * 8 + i;   // node
                int col = n0 + wn * 32 + sn * 16 + r;            // ic
                R[((size_t)b * N_ + row) * IC_ + col] = f2h_bits(fmaxf(acc[sm][sn][i], 0.f));
            }
}

// y[c][node] = Wout(c,:) dot R(node,:) + bout[c]; plus per-channel partial stats
__global__ void k_out(const unsigned short* __restrict__ WoutH,
                      const unsigned short* __restrict__ R,
                      const float* __restrict__ bout,
                      float* __restrict__ y,
                      float* __restrict__ psum, float* __restrict__ psq) {
    __shared__ __align__(16) unsigned short As[2 * BM * LDT];
    __shared__ __align__(16) unsigned short Bs[2 * BN * LDT];
    __shared__ float ssum[BM], ssq[BM];
    int b = blockIdx.z;
    int m0 = blockIdx.y * BM;   // channel
    int n0 = blockIdx.x * BN;   // node
    v8f acc[2][2] = {};
    gemm128x64(WoutH, IC_, R + (size_t)b * N_ * IC_, IC_, IC_, m0, n0, As, Bs, acc);

    if (threadIdx.x < BM) { ssum[threadIdx.x] = 0.f; ssq[threadIdx.x] = 0.f; }
    __syncthreads();

    int lane = threadIdx.x & 31, wid = threadIdx.x >> 5;
    int wm = wid >> 1, wn = wid & 1, r = lane & 15, hf = lane >> 4;
#pragma unroll
    for (int sm = 0; sm < 2; ++sm)
#pragma unroll
        for (int sn = 0; sn < 2; ++sn)
#pragma unroll
            for (int i = 0; i < 8; ++i) {
                int row = m0 + wm * 32 + sm * 16 + hf * 8 + i;   // channel
                int col = n0 + wn * 32 + sn * 16 + r;            // node
                float v = acc[sm][sn][i] + bout[row];
                y[((size_t)b * C_ + row) * N_ + col] = v;
                atomicAdd(&ssum[row - m0], v);
                atomicAdd(&ssq[row - m0], v * v);
            }
    __syncthreads();
    if (threadIdx.x < BM) {
        int c = m0 + threadIdx.x;
        int pb = b * 64 + blockIdx.x;          // 0..127
        psum[c * 128 + pb] = ssum[threadIdx.x];
        psq [c * 128 + pb] = ssq [threadIdx.x];
    }
}

__global__ void k_bnstats(const float* __restrict__ psum, const float* __restrict__ psq,
                          float* __restrict__ mean, float* __restrict__ rstd) {
    int c = threadIdx.x;   // 256 threads, 1 block
    float s = 0.f, q = 0.f;
    for (int i = 0; i < 128; ++i) { s += psum[c * 128 + i]; q += psq[c * 128 + i]; }
    float m = s / 8192.f;
    float var = q / 8192.f - m * m;
    mean[c] = m;
    rstd[c] = rsqrtf(var + 1e-5f);
}

__global__ void k_final(const float* __restrict__ x, const float* __restrict__ y,
                        const float* __restrict__ gamma, const float* __restrict__ beta,
                        const float* __restrict__ mean, const float* __restrict__ rstd,
                        float* __restrict__ out) {
    int idx = blockIdx.x * blockDim.x + threadIdx.x;
    if (idx >= B_ * C_ * N_) return;
    int c = (idx >> 12) & 255;
    out[idx] = x[idx] + gamma[c] * ((y[idx] - mean[c]) * rstd[c]) + beta[c];
}

// ---------------- launcher ----------------

extern "C" void kernel_launch(void* const* d_in, const int* in_sizes, int n_in,
                              void* d_out, int out_size, void* d_ws, size_t ws_size,
                              hipStream_t stream) {
    const float* x        = (const float*)d_in[0];
    const float* hard_map = (const float*)d_in[1];
    const float* Wg       = (const float*)d_in[2];
    const float* bg       = (const float*)d_in[3];
    const float* Wth      = (const float*)d_in[4];
    const float* bth      = (const float*)d_in[5];
    const float* Wph      = (const float*)d_in[6];
    const float* bph      = (const float*)d_in[7];
    const float* Wgcn     = (const float*)d_in[8];
    const float* Wout     = (const float*)d_in[9];
    const float* bout     = (const float*)d_in[10];
    const float* gamma    = (const float*)d_in[11];
    const float* beta     = (const float*)d_in[12];
    float* out = (float*)d_out;

    char* w = (char*)d_ws;
    size_t off = 0;
    auto alloc = [&](size_t bytes) -> void* {
        void* p = w + off;
        off = (off + bytes + 255) & ~(size_t)255;
        return p;
    };
    unsigned short* xT     = (unsigned short*)alloc((size_t)B_ * N_ * C_ * 2);
    unsigned short* thetaN = (unsigned short*)alloc((size_t)B_ * N_ * IC_ * 2);
    unsigned short* phiN   = (unsigned short*)alloc((size_t)B_ * N_ * IC_ * 2);
    unsigned short* gT     = (unsigned short*)alloc((size_t)B_ * IC_ * N_ * 2);
    unsigned short* R      = (unsigned short*)alloc((size_t)B_ * N_ * IC_ * 2);
    unsigned short* att    = (unsigned short*)alloc((size_t)B_ * N_ * N_ * 2);   // 64 MB
    float* y               = (float*)alloc((size_t)B_ * C_ * N_ * 4);            // 8 MB
    unsigned short* WthH   = (unsigned short*)alloc((size_t)IC_ * C_ * 2);
    unsigned short* WphH   = (unsigned short*)alloc((size_t)IC_ * C_ * 2);
    unsigned short* WcombH = (unsigned short*)alloc((size_t)IC_ * C_ * 2);
    unsigned short* WoutH  = (unsigned short*)alloc((size_t)C_ * IC_ * 2);
    float* bcomb           = (float*)alloc(IC_ * 4);
    float* psum            = (float*)alloc((size_t)C_ * 128 * 4);
    float* psq             = (float*)alloc((size_t)C_ * 128 * 4);
    float* meanb           = (float*)alloc(C_ * 4);
    float* rstdb           = (float*)alloc(C_ * 4);

    // weight prep
    k_convert<<<(IC_ * C_ + 255) / 256, 256, 0, stream>>>(Wth, WthH, IC_ * C_);
    k_convert<<<(IC_ * C_ + 255) / 256, 256, 0, stream>>>(Wph, WphH, IC_ * C_);
    k_convert<<<(C_ * IC_ + 255) / 256, 256, 0, stream>>>(Wout, WoutH, C_ * IC_);
    k_wcomb<<<(IC_ * C_) / 256, 256, 0, stream>>>(Wg, Wgcn, bg, WcombH, bcomb);

    // activations
    k_transpose<<<dim3(N_ / 32, C_ / 32, B_), 256, 0, stream>>>(x, xT);
    k_proj<<<dim3(IC_ / BN, N_ / BM, B_ * 3), 256, 0, stream>>>(
        xT, WthH, WphH, WcombH, bth, bph, bcomb, thetaN, phiN, gT);
    k_att<<<dim3(N_ / BN, N_ / BM, B_), 256, 0, stream>>>(thetaN, phiN, att);
    k_select<<<dim3(N_, B_), 256, 0, stream>>>(att, hard_map);
    k_gcn<<<dim3(IC_ / BN, N_ / BM, B_), 256, 0, stream>>>(att, gT, R);
    k_out<<<dim3(N_ / BN, C_ / BM, B_), 256, 0, stream>>>(WoutH, R, bout, y, psum, psq);
    k_bnstats<<<1, 256, 0, stream>>>(psum, psq, meanb, rstdb);
    k_final<<<(B_ * C_ * N_ + 255) / 256, 256, 0, stream>>>(x, y, gamma, beta, meanb, rstdb, out);
}
